// TransformerBlock_70849780515546
// MI455X (gfx1250) — matrix operand order repaired and checked by reading, not compiled, not run on hardware
//
#include <hip/hip_runtime.h>
#include <hip/hip_bf16.h>

typedef _Float16 h16;
typedef __attribute__((ext_vector_type(16))) _Float16 v16h;
typedef __attribute__((ext_vector_type(8)))  float    v8f;

#define DEV static __device__ __forceinline__

// ---------------- WMMA helper: D = A(16x32 f16) * B(32x16 f16) + C(f32) -----
DEV v8f wmma_f16(v16h a, v16h b, v8f c) {
  return __builtin_amdgcn_wmma_f32_16x16x32_f16(
      /*neg_a=*/false, a, /*neg_b=*/false, b,
      /*c_mod=*/(short)0, c, /*reuse_a=*/false, /*reuse_b=*/false);
}

// ---------------- A-fragment offset (16-bit A, 16x32 per k-step) ------------
// Forward layout (ISA 7.12.2): lanes 0-15 hold M=0..15; VGPR v holds
// K = (v&3)*2 + lsb + (v>>2)*16 + (lane>=16 ? 8 : 0).  We store A tiles in
// LDS in exactly per-lane order: offset = ((kstep*32 + lane)*16 + e) halves,
// so A loads are two contiguous ds_load_b128 per lane.
DEV int a_off(int m, int k) {           // k in [0,128)
  int s  = k >> 5;
  int kk = k & 31;
  int hi = (kk >> 3) & 1;               // which half-wave group
  int t  = kk - (hi << 3);              // t in {0..7} u {16..23}
  int v  = (t >= 16) ? (4 + ((t - 16) >> 1)) : (t >> 1);
  int e  = (v << 1) | (t & 1);
  int L  = (m & 15) + (hi << 4);
  return ((s * 32 + L) << 4) + e;       // units: halves
}

// ---------------- K0: convert f32 weight [Din][Dout] -> f16 B-fragments -----
// B layout (32x16 f16): lane L holds column (L&15); halves e cover
// K = (e>=8 ? 16 : 0) + (L>=16 ? 8 : 0) + (e&7) within the 32-deep k-step.
// Stored as [kt][ct][lane][e] so each wave reads its fragment contiguously.
__global__ __launch_bounds__(256) void wconv_kernel(const float* __restrict__ W,
                                                    h16* __restrict__ Wp,
                                                    int Din, int Dout) {
  int off = blockIdx.x * 256 + threadIdx.x;
  if (off >= Din * Dout) return;
  int nc   = Dout >> 4;
  int e    = off & 15;
  int L    = (off >> 4) & 31;
  int tile = off >> 9;
  int ct   = tile % nc;
  int kt   = tile / nc;
  int k    = kt * 32 + ((e >> 3) << 4) + ((L >> 4) << 3) + (e & 7);
  int c    = (ct << 4) + (L & 15);
  Wp[off]  = (h16)W[k * Dout + c];
}

// ---------------- K1: brute-force KNN, top-16 by insertion in registers -----
__global__ __launch_bounds__(256) void knn_kernel(const float* __restrict__ x,
                                                  int* __restrict__ knn,
                                                  int N) {
  __shared__ float sx[256 * 3];
  int blocksPerBatch = N / 256;
  int b  = blockIdx.x / blocksPerBatch;
  int n  = (blockIdx.x % blocksPerBatch) * 256 + threadIdx.x;
  const float* xb = x + (size_t)b * N * 3;
  float xi0 = xb[n * 3 + 0], xi1 = xb[n * 3 + 1], xi2 = xb[n * 3 + 2];

  float bd[16]; int bi[16];
#pragma unroll
  for (int i = 0; i < 16; ++i) { bd[i] = 3.0e38f; bi[i] = 0; }

  for (int tile = 0; tile < N; tile += 256) {
    __syncthreads();
    int j = tile + threadIdx.x;
    sx[threadIdx.x * 3 + 0] = xb[j * 3 + 0];
    sx[threadIdx.x * 3 + 1] = xb[j * 3 + 1];
    sx[threadIdx.x * 3 + 2] = xb[j * 3 + 2];
    __syncthreads();
    for (int t = 0; t < 256; ++t) {
      float d0 = xi0 - sx[t * 3 + 0];
      float d1 = xi1 - sx[t * 3 + 1];
      float d2 = xi2 - sx[t * 3 + 2];
      float v  = d0 * d0 + d1 * d1 + d2 * d2;
      if (v < bd[15]) {
        bd[15] = v; bi[15] = tile + t;
#pragma unroll
        for (int i = 15; i > 0; --i) {
          if (bd[i] < bd[i - 1]) {
            float tf = bd[i]; bd[i] = bd[i - 1]; bd[i - 1] = tf;
            int   ti = bi[i]; bi[i] = bi[i - 1]; bi[i - 1] = ti;
          }
        }
      }
    }
  }
  size_t base = ((size_t)b * N + n) * 16;
#pragma unroll
  for (int i = 0; i < 16; ++i) knn[base + i] = bi[i];
}

// ---------------- K2: f = in_f@Wfc1+b; psi/alpha/phi; u = phi + dptMLP(f) ---
// Block = 16 points x 8 waves (one 16-col tile each). All GEMMs via WMMA.
__global__ __launch_bounds__(256) void feat_kernel(
    const float* __restrict__ in_f,
    const float* __restrict__ b_fc1, const float* __restrict__ b_dpt1,
    const float* __restrict__ b_dpt2,
    const h16* __restrict__ Wfc1p, const h16* __restrict__ Wphip,
    const h16* __restrict__ Wpsip, const h16* __restrict__ Walphap,
    const h16* __restrict__ Wdpt1p, const h16* __restrict__ Wdpt2p,
    h16* __restrict__ psi16, h16* __restrict__ alpha16, h16* __restrict__ u16) {
  __shared__ h16 S_f[4 * 32 * 16];   // f A-fragments (K=128 -> 4 ksteps)
  __shared__ h16 S_h[4 * 32 * 16];   // in_f frags, later dpt1 hidden frags

  int lane = threadIdx.x & 31, wave = threadIdx.x >> 5;
  size_t row0 = (size_t)blockIdx.x * 16;   // flat point base (b*N+n)

  // stage in_f (16x64) as A-fragments (2 ksteps)
  for (int i = threadIdx.x; i < 16 * 64; i += 256) {
    int m = i >> 6, k = i & 63;
    S_h[a_off(m, k)] = (h16)in_f[(row0 + m) * 64 + k];
  }
  __syncthreads();

  int c = lane & 15, hi = lane >> 4, d = wave * 16 + c;

  // fc1
  v8f f;
  { float bb = b_fc1[d];
#pragma unroll
    for (int r = 0; r < 8; ++r) f[r] = bb; }
#pragma unroll
  for (int s = 0; s < 2; ++s) {
    v16h a = *(const v16h*)&S_h[(s * 32 + lane) * 16];
    v16h w = *(const v16h*)&Wfc1p[((s * 8 + wave) * 32 + lane) * 16];
    f = wmma_f16(a, w, f);
  }
#pragma unroll
  for (int r = 0; r < 8; ++r) S_f[a_off(r + 8 * hi, d)] = (h16)f[r];
  __syncthreads();     // S_f ready; S_h free

  // psi / alpha / phi (share A fragments)
  v8f ps, al, ph;
#pragma unroll
  for (int r = 0; r < 8; ++r) { ps[r] = 0.f; al[r] = 0.f; ph[r] = 0.f; }
#pragma unroll
  for (int s = 0; s < 4; ++s) {
    v16h a = *(const v16h*)&S_f[(s * 32 + lane) * 16];
    ps = wmma_f16(a, *(const v16h*)&Wpsip  [((s * 8 + wave) * 32 + lane) * 16], ps);
    al = wmma_f16(a, *(const v16h*)&Walphap[((s * 8 + wave) * 32 + lane) * 16], al);
    ph = wmma_f16(a, *(const v16h*)&Wphip  [((s * 8 + wave) * 32 + lane) * 16], ph);
  }
#pragma unroll
  for (int r = 0; r < 8; ++r) {
    size_t row = row0 + r + 8 * hi;
    psi16  [row * 128 + d] = (h16)ps[r];
    alpha16[row * 128 + d] = (h16)al[r];
  }

  // dpt1 (relu) -> S_h fragments
  v8f h1;
  { float bb = b_dpt1[d];
#pragma unroll
    for (int r = 0; r < 8; ++r) h1[r] = bb; }
#pragma unroll
  for (int s = 0; s < 4; ++s) {
    v16h a = *(const v16h*)&S_f[(s * 32 + lane) * 16];
    h1 = wmma_f16(a, *(const v16h*)&Wdpt1p[((s * 8 + wave) * 32 + lane) * 16], h1);
  }
#pragma unroll
  for (int r = 0; r < 8; ++r)
    S_h[a_off(r + 8 * hi, d)] = (h16)fmaxf(h1[r], 0.f);
  __syncthreads();

  // dpt2 + phi -> u
  v8f u;
  { float bb = b_dpt2[d];
#pragma unroll
    for (int r = 0; r < 8; ++r) u[r] = bb; }
#pragma unroll
  for (int s = 0; s < 4; ++s) {
    v16h a = *(const v16h*)&S_h[(s * 32 + lane) * 16];
    u = wmma_f16(a, *(const v16h*)&Wdpt2p[((s * 8 + wave) * 32 + lane) * 16], u);
  }
#pragma unroll
  for (int r = 0; r < 8; ++r) {
    size_t row = row0 + r + 8 * hi;
    u16[row * 128 + d] = (h16)(u[r] + ph[r]);
  }
}

// ---------------- K3: per-point attention core (2 points per block) ---------
__global__ __launch_bounds__(256) void attn_kernel(
    const float* __restrict__ x, const int* __restrict__ knn,
    const float* __restrict__ Wd1, const float* __restrict__ bd1,
    const float* __restrict__ bd2, const float* __restrict__ bg1,
    const float* __restrict__ bg2,
    const h16* __restrict__ Wd2p, const h16* __restrict__ Wg1p,
    const h16* __restrict__ Wg2p,
    const h16* __restrict__ psi16, const h16* __restrict__ alpha16,
    const h16* __restrict__ u16, h16* __restrict__ y16, int N) {
  __shared__ h16   S_psi[2 * 16 * 128];    // row-major gathered psi
  __shared__ h16   S_alpha[2 * 16 * 128];  // row-major gathered alpha
  __shared__ h16   S_ka[2 * 4 * 32 * 16];  // frag buf: deltaH then gam1-hidden
  __shared__ h16   S_pre[2 * 4 * 32 * 16]; // pre fragments
  __shared__ float S_g[8 * 16 * 16];       // per-wave gamma scratch (col-major)
  __shared__ float S_dl[8 * 16 * 16];      // per-wave delta scratch (col-major)
  __shared__ h16   S_u[2 * 128];
  __shared__ float S_diff[2 * 16 * 3];
  __shared__ int   S_idx[2 * 16];
  __shared__ float S_w1[3 * 128 + 128];    // W_del1 + b_del1

  int lane = threadIdx.x & 31, wave = threadIdx.x >> 5;
  size_t pt0  = (size_t)blockIdx.x * 2;
  int    b    = (int)(pt0 / N);
  size_t brow = (size_t)b * N;

  for (int i = threadIdx.x; i < 512; i += 256)
    S_w1[i] = (i < 384) ? Wd1[i] : bd1[i - 384];
  if (threadIdx.x < 32)
    S_idx[threadIdx.x] = knn[(pt0 + (threadIdx.x >> 4)) * 16 + (threadIdx.x & 15)];
  if (threadIdx.x >= 32 && threadIdx.x < 64) {
    int t = threadIdx.x - 32;                       // 32 x uint4 = 2x128 halves
    ((uint4*)S_u)[t] = ((const uint4*)(u16 + pt0 * 128))[t];
  }
  __syncthreads();

  if (threadIdx.x < 32) {
    int p = threadIdx.x >> 4, m = threadIdx.x & 15;
    int j = S_idx[p * 16 + m];
    const float* xi = x + (pt0 + p) * 3;
    const float* xj = x + (brow + j) * 3;
#pragma unroll
    for (int cc = 0; cc < 3; ++cc)
      S_diff[(p * 16 + m) * 3 + cc] = xi[cc] - xj[cc];
  }
  // gather psi/alpha rows (16B chunks)
  for (int i = threadIdx.x; i < 512; i += 256) {
    int row = i >> 4, ch = i & 15;
    int j   = S_idx[row];
    size_t src = (brow + j) * 128 + (size_t)ch * 8;
    ((uint4*)S_psi)[i]   = *(const uint4*)(psi16 + src);
    ((uint4*)S_alpha)[i] = *(const uint4*)(alpha16 + src);
  }
  __syncthreads();

  // deltaH = relu(diff @ W_del1 + b_del1), stored as A-fragments
  for (int i = threadIdx.x; i < 4096; i += 256) {
    int d = i & 127, row = i >> 7, p = row >> 4, m = row & 15;
    float a = S_diff[row * 3 + 0] * S_w1[d]
            + S_diff[row * 3 + 1] * S_w1[128 + d]
            + S_diff[row * 3 + 2] * S_w1[256 + d] + S_w1[384 + d];
    S_ka[p * 2048 + a_off(m, d)] = (h16)fmaxf(a, 0.f);
  }
  __syncthreads();

  int c = lane & 15, hi = lane >> 4, d = wave * 16 + c;
  v16h bf[4];
  v8f  delta[2];

  // --- del2: delta = deltaH @ W_del2 + b_del2; fuse pre = u - psi + delta ---
#pragma unroll
  for (int s = 0; s < 4; ++s)
    bf[s] = *(const v16h*)&Wd2p[((s * 8 + wave) * 32 + lane) * 16];
  {
    float bb = bd2[d];
#pragma unroll
    for (int p = 0; p < 2; ++p) {
      v8f acc;
#pragma unroll
      for (int r = 0; r < 8; ++r) acc[r] = bb;
#pragma unroll
      for (int s = 0; s < 4; ++s)
        acc = wmma_f16(*(const v16h*)&S_ka[p * 2048 + (s * 32 + lane) * 16],
                       bf[s], acc);
      delta[p] = acc;
      float uv = (float)S_u[p * 128 + d];
#pragma unroll
      for (int r = 0; r < 8; ++r) {
        int m = r + 8 * hi;
        float pv = uv - (float)S_psi[(p * 16 + m) * 128 + d] + acc[r];
        S_pre[p * 2048 + a_off(m, d)] = (h16)pv;
      }
    }
  }
  __syncthreads();

  // --- gam1 (relu) -> S_ka fragments ---
#pragma unroll
  for (int s = 0; s < 4; ++s)
    bf[s] = *(const v16h*)&Wg1p[((s * 8 + wave) * 32 + lane) * 16];
  {
    float bb = bg1[d];
#pragma unroll
    for (int p = 0; p < 2; ++p) {
      v8f acc;
#pragma unroll
      for (int r = 0; r < 8; ++r) acc[r] = bb;
#pragma unroll
      for (int s = 0; s < 4; ++s)
        acc = wmma_f16(*(const v16h*)&S_pre[p * 2048 + (s * 32 + lane) * 16],
                       bf[s], acc);
#pragma unroll
      for (int r = 0; r < 8; ++r)
        S_ka[p * 2048 + a_off(r + 8 * hi, d)] = (h16)fmaxf(acc[r], 0.f);
    }
  }
  __syncthreads();

  // --- gam2 + softmax over k (the 16 rows) + weighted sum -> y ---
#pragma unroll
  for (int s = 0; s < 4; ++s)
    bf[s] = *(const v16h*)&Wg2p[((s * 8 + wave) * 32 + lane) * 16];
  float  bb2 = bg2[d];
  float* gsc = S_g  + wave * 256;
  float* dsc = S_dl + wave * 256;
  for (int p = 0; p < 2; ++p) {
    v8f g;
#pragma unroll
    for (int r = 0; r < 8; ++r) g[r] = bb2;
#pragma unroll
    for (int s = 0; s < 4; ++s)
      g = wmma_f16(*(const v16h*)&S_ka[p * 2048 + (s * 32 + lane) * 16],
                   bf[s], g);
#pragma unroll
    for (int r = 0; r < 8; ++r) {
      int m = r + 8 * hi;
      gsc[c * 16 + m] = g[r] * 0.25f;          // / sqrt(K), K=16
      dsc[c * 16 + m] = delta[p][r];
    }
    __syncthreads();
    if (lane < 16) {                           // one lane per column d
      float mx = -3.0e38f;
#pragma unroll
      for (int kk = 0; kk < 16; ++kk) mx = fmaxf(mx, gsc[lane * 16 + kk]);
      float e[16], sum = 0.f;
#pragma unroll
      for (int kk = 0; kk < 16; ++kk) {
        e[kk] = __expf(gsc[lane * 16 + kk] - mx);
        sum += e[kk];
      }
      float inv = 1.f / sum, y = 0.f;
#pragma unroll
      for (int kk = 0; kk < 16; ++kk)
        y += e[kk] * inv *
             ((float)S_alpha[(p * 16 + kk) * 128 + d] + dsc[lane * 16 + kk]);
      y16[(pt0 + p) * 128 + d] = (h16)y;
    }
    __syncthreads();
  }
}

// ---------------- K4: out = y @ W_fc2 + b_fc2 + in_f ------------------------
__global__ __launch_bounds__(256) void out_kernel(
    const h16* __restrict__ y16, const h16* __restrict__ Wfc2p,
    const float* __restrict__ b_fc2, const float* __restrict__ in_f,
    float* __restrict__ out) {
  __shared__ h16 S_a[2 * 4 * 32 * 16];       // two 16-row groups of A frags
  int lane = threadIdx.x & 31, wave = threadIdx.x >> 5;
  int grp = wave >> 2, wc = wave & 3;
  size_t row0 = (size_t)blockIdx.x * 32;

  for (int i = threadIdx.x; i < 32 * 128; i += 256) {
    int k = i & 127, row = i >> 7, g = row >> 4, m = row & 15;
    S_a[g * 2048 + a_off(m, k)] = y16[(row0 + row) * 128 + k];
  }
  __syncthreads();

  int c = lane & 15, hi = lane >> 4, d = wc * 16 + c;
  v8f acc;
  { float bb = b_fc2[d];
#pragma unroll
    for (int r = 0; r < 8; ++r) acc[r] = bb; }
#pragma unroll
  for (int s = 0; s < 4; ++s)
    acc = wmma_f16(*(const v16h*)&S_a[grp * 2048 + (s * 32 + lane) * 16],
                   *(const v16h*)&Wfc2p[((s * 4 + wc) * 32 + lane) * 16], acc);
#pragma unroll
  for (int r = 0; r < 8; ++r) {
    size_t row = row0 + grp * 16 + r + 8 * hi;
    out[row * 64 + d] = acc[r] + in_f[row * 64 + d];
  }
}

// ---------------- host-side launcher ----------------------------------------
extern "C" void kernel_launch(void* const* d_in, const int* in_sizes, int n_in,
                              void* d_out, int out_size, void* d_ws, size_t ws_size,
                              hipStream_t stream) {
  constexpr int B = 8, N = 2048, BN = B * N;

  const float* x      = (const float*)d_in[0];
  const float* in_f   = (const float*)d_in[1];
  const float* W_fc1  = (const float*)d_in[2];
  const float* b_fc1  = (const float*)d_in[3];
  const float* W_phi  = (const float*)d_in[4];
  const float* W_psi  = (const float*)d_in[5];
  const float* W_alpha= (const float*)d_in[6];
  const float* W_dpt1 = (const float*)d_in[7];
  const float* b_dpt1 = (const float*)d_in[8];
  const float* W_dpt2 = (const float*)d_in[9];
  const float* b_dpt2 = (const float*)d_in[10];
  const float* W_gam1 = (const float*)d_in[11];
  const float* b_gam1 = (const float*)d_in[12];
  const float* W_gam2 = (const float*)d_in[13];
  const float* b_gam2 = (const float*)d_in[14];
  const float* W_del1 = (const float*)d_in[15];
  const float* b_del1 = (const float*)d_in[16];
  const float* W_del2 = (const float*)d_in[17];
  const float* b_del2 = (const float*)d_in[18];
  const float* W_fc2  = (const float*)d_in[19];
  const float* b_fc2  = (const float*)d_in[20];
  float* out = (float*)d_out;

  // workspace carve-up
  char* p = (char*)d_ws;
  auto take = [&](size_t bytes) { char* r = p; p += (bytes + 255) & ~size_t(255); return r; };
  int* knn      = (int*)take((size_t)BN * 16 * sizeof(int));
  h16* psi16    = (h16*)take((size_t)BN * 128 * 2);
  h16* alpha16  = (h16*)take((size_t)BN * 128 * 2);
  h16* u16      = (h16*)take((size_t)BN * 128 * 2);
  h16* y16      = (h16*)take((size_t)BN * 128 * 2);
  h16* Wfc1p    = (h16*)take(64  * 128 * 2);
  h16* Wphip    = (h16*)take(128 * 128 * 2);
  h16* Wpsip    = (h16*)take(128 * 128 * 2);
  h16* Walphap  = (h16*)take(128 * 128 * 2);
  h16* Wdpt1p   = (h16*)take(128 * 128 * 2);
  h16* Wdpt2p   = (h16*)take(128 * 128 * 2);
  h16* Wd2p     = (h16*)take(128 * 128 * 2);
  h16* Wg1p     = (h16*)take(128 * 128 * 2);
  h16* Wg2p     = (h16*)take(128 * 128 * 2);
  h16* Wfc2p    = (h16*)take(128 * 64  * 2);

  auto conv = [&](const float* W, h16* Wp, int din, int dout) {
    int total = din * dout;
    wconv_kernel<<<(total + 255) / 256, 256, 0, stream>>>(W, Wp, din, dout);
  };
  conv(W_fc1,  Wfc1p,  64, 128);
  conv(W_phi,  Wphip,  128, 128);
  conv(W_psi,  Wpsip,  128, 128);
  conv(W_alpha,Walphap,128, 128);
  conv(W_dpt1, Wdpt1p, 128, 128);
  conv(W_dpt2, Wdpt2p, 128, 128);
  conv(W_del2, Wd2p,   128, 128);
  conv(W_gam1, Wg1p,   128, 128);
  conv(W_gam2, Wg2p,   128, 128);
  conv(W_fc2,  Wfc2p,  128, 64);

  knn_kernel<<<B * (N / 256), 256, 0, stream>>>(x, knn, N);

  feat_kernel<<<BN / 16, 256, 0, stream>>>(
      in_f, b_fc1, b_dpt1, b_dpt2,
      Wfc1p, Wphip, Wpsip, Walphap, Wdpt1p, Wdpt2p,
      psi16, alpha16, u16);

  attn_kernel<<<BN / 2, 256, 0, stream>>>(
      x, knn, W_del1, b_del1, b_del2, b_gam1, b_gam2,
      Wd2p, Wg1p, Wg2p, psi16, alpha16, u16, y16, N);

  out_kernel<<<BN / 32, 256, 0, stream>>>(y16, Wfc2p, b_fc2, in_f, out);
}
